// Model_39676907883791
// MI455X (gfx1250) — matrix operand order repaired and checked by reading, not compiled, not run on hardware
//
#include <hip/hip_runtime.h>

typedef __attribute__((ext_vector_type(16))) _Float16 v16h;
typedef __attribute__((ext_vector_type(8)))  float    v8f;

#define SEQ   1024
#define HDIM  64
#define KTILE 64
#define QTILE 64
#define NWAVE 4
#define NT    (SEQ / KTILE)
#define SCALE 10000.0f
#define KEEP_THRESH 3865470566u   /* floor(0.9 * 2^32) */
#define INV_KEEP    (1.0f / 0.9f)

__device__ __forceinline__ unsigned rng_hash(unsigned x) {
    x ^= 42u * 0x9e3779b9u;
    x ^= x >> 16; x *= 0x7feb352dU;
    x ^= x >> 15; x *= 0x846ca68bU;
    x ^= x >> 16;
    return x;
}

// Low 32 bits of a generic pointer into LDS == LDS byte offset (flat aperture rule).
__device__ __forceinline__ unsigned lds_off(const void* p) {
    return (unsigned)(unsigned long long)p;
}

// ---------------------------------------------------------------------------
// Shared per-tile compute: scores WMMA -> online softmax+dropout -> PV WMMA.
// Kl: [key][d] f16 row-major (KTILE x HDIM); Vt: [d][key] f16 (HDIM x KTILE).
// ---------------------------------------------------------------------------
__device__ __forceinline__ void attn_tile(
    const _Float16* __restrict__ Kl, const _Float16* __restrict__ Vt,
    _Float16* __restrict__ PstW,
    const v16h (&aq)[2], v8f (&acc)[4], float (&mrow)[8], float (&lrow)[8],
    int k0, int col, bool hiL, int loOff, int bOff, int wave, int bh, int q0)
{
    const v8f vzero = {0.f, 0.f, 0.f, 0.f, 0.f, 0.f, 0.f, 0.f};

    // ---- Scores: S(16x64) = Q(16x64) . K^T ; 8 WMMAs ----
    v8f sc[4] = {vzero, vzero, vzero, vzero};
    #pragma unroll
    for (int c = 0; c < 2; ++c) {
        #pragma unroll
        for (int n = 0; n < 4; ++n) {
            const v16h bk = *(const v16h*)&Kl[(n * 16 + col) * HDIM + c * 32 + bOff];
            sc[n] = __builtin_amdgcn_wmma_f32_16x16x32_f16(
                false, aq[c], false, bk, (short)0, sc[n], false, false);
        }
    }

    // ---- Online softmax (rows j -> global row j + 8*hiL) ----
    float mnew[8], resc[8], tsum[8];
    #pragma unroll
    for (int j = 0; j < 8; ++j) {
        float mx = -1e30f;
        #pragma unroll
        for (int n = 0; n < 4; ++n) mx = fmaxf(mx, sc[n][j] * SCALE);
        #pragma unroll
        for (int mk = 1; mk < 16; mk <<= 1)
            mx = fmaxf(mx, __shfl_xor(mx, mk, 32));
        mnew[j] = fmaxf(mrow[j], mx);
        resc[j] = __expf(mrow[j] - mnew[j]);
        mrow[j] = mnew[j];
        tsum[j] = 0.f;
    }

    // exp, pre-dropout l-sum, dropout, stage P into per-wave LDS
    const unsigned qbase = (unsigned)(q0 + wave * 16 + (hiL ? 8 : 0));
    #pragma unroll
    for (int n = 0; n < 4; ++n) {
        const unsigned kglob = (unsigned)(k0 + n * 16 + col);
        #pragma unroll
        for (int j = 0; j < 8; ++j) {
            const float p = __expf(sc[n][j] * SCALE - mnew[j]);
            tsum[j] += p;
            const unsigned h =
                rng_hash(((unsigned)bh * SEQ + (qbase + j)) * SEQ + kglob);
            const float pd = (h < KEEP_THRESH) ? p * INV_KEEP : 0.f;
            PstW[(j + (hiL ? 8 : 0)) * KTILE + n * 16 + col] = (_Float16)pd;
        }
    }
    #pragma unroll
    for (int j = 0; j < 8; ++j) {
        #pragma unroll
        for (int mk = 1; mk < 16; mk <<= 1)
            tsum[j] += __shfl_xor(tsum[j], mk, 32);
        lrow[j] = lrow[j] * resc[j] + tsum[j];
    }
    #pragma unroll
    for (int n = 0; n < 4; ++n)
        #pragma unroll
        for (int j = 0; j < 8; ++j)
            acc[n][j] *= resc[j];

    // ---- Reload P as A-fragments (wave-private LDS: DS ops in-order per wave) ----
    v16h pa[2];
    #pragma unroll
    for (int c = 0; c < 2; ++c) {
        const _Float16* prow = &PstW[col * KTILE + c * 32 + bOff];
        #pragma unroll
        for (int i = 0; i < 8; ++i) {
            pa[c][i]     = prow[i];
            pa[c][8 + i] = prow[8 + i];
        }
    }

    // ---- acc(16x64) += P(16x64) . V(64x64) ; 8 WMMAs ----
    #pragma unroll
    for (int n = 0; n < 4; ++n) {
        #pragma unroll
        for (int c = 0; c < 2; ++c) {
            const v16h bv = *(const v16h*)&Vt[(n * 16 + col) * KTILE + c * 32 + bOff];
            acc[n] = __builtin_amdgcn_wmma_f32_16x16x32_f16(
                false, pa[c], false, bv, (short)0, acc[n], false, false);
        }
    }
}

// ---------------------------------------------------------------------------
// Pass 1: one-shot fp32 -> f16 conversion of K (row-major) and V (transposed
// per (b,h): Vt[bh][d][s]).  Amortizes conversion across the 16 q-tile blocks
// that reuse each (b,h) slice and makes tiles raw byte-copyable for the async
// LDS DMA path.
// ---------------------------------------------------------------------------
__global__ __launch_bounds__(256)
void convert_kv(const float* __restrict__ K, const float* __restrict__ V,
                _Float16* __restrict__ Kh, _Float16* __restrict__ Vt)
{
    const size_t base = ((size_t)blockIdx.x * 256 + threadIdx.x) * 4;

    const float4 k4 = *(const float4*)(K + base);
    _Float16* kd = Kh + base;
    kd[0] = (_Float16)k4.x; kd[1] = (_Float16)k4.y;
    kd[2] = (_Float16)k4.z; kd[3] = (_Float16)k4.w;

    const float4 v4 = *(const float4*)(V + base);
    const int    d  = (int)(base & 63);
    const size_t s  = (base >> 6) & (SEQ - 1);
    const size_t bh = base >> 16;
    _Float16* vt = Vt + (bh * HDIM) * SEQ + s;
    vt[(size_t)(d + 0) * SEQ] = (_Float16)v4.x;
    vt[(size_t)(d + 1) * SEQ] = (_Float16)v4.y;
    vt[(size_t)(d + 2) * SEQ] = (_Float16)v4.z;
    vt[(size_t)(d + 3) * SEQ] = (_Float16)v4.w;
}

// Issue one K+V tile as 8 async b128 copies per wave (ASYNCcnt += 8).
__device__ __forceinline__ void issue_tile_async(
    const _Float16* __restrict__ Khb,   // (b,h) slice, [s][d] f16
    const _Float16* __restrict__ Vtb,   // (b,h) slice, [d][s] f16
    int k0, _Float16* klbuf, _Float16* vtbuf, int tid)
{
    // K tile: contiguous 8 KB block; 4 x (128 lanes x 16 B)
    const void* kbase = (const void*)(Khb + (size_t)k0 * HDIM);
    #pragma unroll
    for (int r = 0; r < 4; ++r) {
        unsigned off = (unsigned)(r * 2048 + tid * 16);
        unsigned l   = lds_off(klbuf) + off;
        asm volatile("global_load_async_to_lds_b128 %0, %1, %2"
                     :: "v"(l), "v"(off), "s"(kbase) : "memory");
    }
    // V tile: 64 rows of 128 B, global row stride 2 KB -> LDS row stride 128 B
    const void* vbase = (const void*)((const char*)Vtb + (size_t)k0 * 2);
    #pragma unroll
    for (int r = 0; r < 4; ++r) {
        const int c = r * 128 + tid;          // 0..511 chunk id
        const int d = c >> 3, i = c & 7;
        unsigned goff = (unsigned)(d * (SEQ * 2) + i * 16);
        unsigned l    = lds_off(vtbuf) + (unsigned)(d * 128 + i * 16);
        asm volatile("global_load_async_to_lds_b128 %0, %1, %2"
                     :: "v"(l), "v"(goff), "s"(vbase) : "memory");
    }
}

// ---------------------------------------------------------------------------
// Pass 2: flash attention, double-buffered async LDS staging.
// ---------------------------------------------------------------------------
__global__ __launch_bounds__(128)
void attn_fwd_async(const float* __restrict__ Q,
                    const _Float16* __restrict__ Kh,
                    const _Float16* __restrict__ Vtg,
                    float* __restrict__ O)
{
    __shared__ alignas(32) _Float16 KlB[2][KTILE * HDIM];
    __shared__ alignas(32) _Float16 VtB[2][HDIM * KTILE];
    __shared__ alignas(32) _Float16 Pst[NWAVE][16 * KTILE];

    const int tid  = threadIdx.x;
    const int wave = tid >> 5;
    const int lane = tid & 31;
    const int col  = lane & 15;
    const bool hiL = lane >= 16;
    const int loOff = hiL ? 8 : 0;
    const int bOff  = hiL ? 16 : 0;

    const int bh    = blockIdx.x >> 4;
    const int qtile = blockIdx.x & 15;
    const int q0    = qtile * QTILE;

    const float*    Qb  = Q   + (size_t)bh * SEQ * HDIM;
    const _Float16* Khb = Kh  + (size_t)bh * SEQ * HDIM;
    const _Float16* Vtb = Vtg + (size_t)bh * HDIM * SEQ;
    float*          Ob  = O   + (size_t)bh * SEQ * HDIM;

    // Kick off tile 0 before anything else: overlap with Q fragment setup.
    issue_tile_async(Khb, Vtb, 0, KlB[0], VtB[0], tid);

    // Q A-fragments in VGPRs for the whole kernel
    v16h aq[2];
    {
        const float* row = Qb + (size_t)(q0 + wave * 16 + col) * HDIM;
        #pragma unroll
        for (int c = 0; c < 2; ++c) {
            const int dbase = c * 32 + bOff;
            #pragma unroll
            for (int i = 0; i < 8; ++i) {
                aq[c][i]     = (_Float16)row[dbase + i];
                aq[c][8 + i] = (_Float16)row[dbase + 8 + i];
            }
        }
    }

    const v8f vzero = {0.f, 0.f, 0.f, 0.f, 0.f, 0.f, 0.f, 0.f};
    v8f acc[4] = {vzero, vzero, vzero, vzero};
    float mrow[8], lrow[8];
    #pragma unroll
    for (int j = 0; j < 8; ++j) { mrow[j] = -1e30f; lrow[j] = 0.f; }

    for (int kt = 0; kt < NT; ++kt) {
        const int cur = kt & 1;
        if (kt + 1 < NT) {
            // Buffer cur^1 was fully consumed by the barrier at the end of the
            // previous iteration -> safe to overwrite via the async engine.
            issue_tile_async(Khb, Vtb, (kt + 1) * KTILE,
                             KlB[cur ^ 1], VtB[cur ^ 1], tid);
            // 16 outstanding; first 8 (tile kt) complete in order.
            asm volatile("s_wait_asynccnt 0x8" ::: "memory");
        } else {
            asm volatile("s_wait_asynccnt 0x0" ::: "memory");
        }
        __syncthreads();   // all waves' tile-kt async copies landed in LDS

        attn_tile(KlB[cur], VtB[cur], Pst[wave],
                  aq, acc, mrow, lrow,
                  kt * KTILE, col, hiL, loOff, bOff, wave, bh, q0);

        __syncthreads();   // readers done before this buffer is re-targeted
    }

    #pragma unroll
    for (int n = 0; n < 4; ++n) {
        #pragma unroll
        for (int j = 0; j < 8; ++j) {
            const int r = q0 + wave * 16 + j + (hiL ? 8 : 0);
            Ob[(size_t)r * HDIM + n * 16 + col] = acc[n][j] / lrow[j];
        }
    }
}

// ---------------------------------------------------------------------------
// Fallback (workspace too small): single-pass kernel, in-loop conversion.
// ---------------------------------------------------------------------------
__global__ __launch_bounds__(128)
void attn_fwd_f16wmma(const float* __restrict__ Q,
                      const float* __restrict__ K,
                      const float* __restrict__ V,
                      float* __restrict__ O)
{
    __shared__ alignas(32) _Float16 Kl[KTILE * HDIM];
    __shared__ alignas(32) _Float16 Vt[HDIM * KTILE];
    __shared__ alignas(32) _Float16 Pst[NWAVE][16 * KTILE];

    const int tid  = threadIdx.x;
    const int wave = tid >> 5;
    const int lane = tid & 31;
    const int col  = lane & 15;
    const bool hiL = lane >= 16;
    const int loOff = hiL ? 8 : 0;
    const int bOff  = hiL ? 16 : 0;

    const int bh    = blockIdx.x >> 4;
    const int qtile = blockIdx.x & 15;
    const int q0    = qtile * QTILE;

    const float* Qb = Q + (size_t)bh * SEQ * HDIM;
    const float* Kb = K + (size_t)bh * SEQ * HDIM;
    const float* Vb = V + (size_t)bh * SEQ * HDIM;
    float*       Ob = O + (size_t)bh * SEQ * HDIM;

    v16h aq[2];
    {
        const float* row = Qb + (size_t)(q0 + wave * 16 + col) * HDIM;
        #pragma unroll
        for (int c = 0; c < 2; ++c) {
            const int dbase = c * 32 + bOff;
            #pragma unroll
            for (int i = 0; i < 8; ++i) {
                aq[c][i]     = (_Float16)row[dbase + i];
                aq[c][8 + i] = (_Float16)row[dbase + 8 + i];
            }
        }
    }

    const v8f vzero = {0.f, 0.f, 0.f, 0.f, 0.f, 0.f, 0.f, 0.f};
    v8f acc[4] = {vzero, vzero, vzero, vzero};
    float mrow[8], lrow[8];
    #pragma unroll
    for (int j = 0; j < 8; ++j) { mrow[j] = -1e30f; lrow[j] = 0.f; }

    for (int kt = 0; kt < NT; ++kt) {
        const int k0 = kt * KTILE;
        if (kt + 1 < NT) {
            __builtin_prefetch(Kb + (size_t)(k0 + KTILE) * HDIM + tid * 4, 0, 1);
            __builtin_prefetch(Vb + (size_t)(k0 + KTILE) * HDIM + tid * 4, 0, 1);
        }
        __syncthreads();
        #pragma unroll
        for (int i = 0; i < 8; ++i) {
            const int idx = i * 512 + tid * 4;
            const float4 kv = *(const float4*)(Kb + (size_t)k0 * HDIM + idx);
            _Float16* kd = &Kl[idx];
            kd[0] = (_Float16)kv.x; kd[1] = (_Float16)kv.y;
            kd[2] = (_Float16)kv.z; kd[3] = (_Float16)kv.w;
            const float4 vv = *(const float4*)(Vb + (size_t)k0 * HDIM + idx);
            const int key = idx >> 6, dd = idx & 63;
            Vt[(dd + 0) * KTILE + key] = (_Float16)vv.x;
            Vt[(dd + 1) * KTILE + key] = (_Float16)vv.y;
            Vt[(dd + 2) * KTILE + key] = (_Float16)vv.z;
            Vt[(dd + 3) * KTILE + key] = (_Float16)vv.w;
        }
        __syncthreads();

        attn_tile(Kl, Vt, Pst[wave], aq, acc, mrow, lrow,
                  k0, col, hiL, loOff, bOff, wave, bh, q0);
    }

    #pragma unroll
    for (int n = 0; n < 4; ++n) {
        #pragma unroll
        for (int j = 0; j < 8; ++j) {
            const int r = q0 + wave * 16 + j + (hiL ? 8 : 0);
            Ob[(size_t)r * HDIM + n * 16 + col] = acc[n][j] / lrow[j];
        }
    }
}

extern "C" void kernel_launch(void* const* d_in, const int* in_sizes, int n_in,
                              void* d_out, int out_size, void* d_ws, size_t ws_size,
                              hipStream_t stream) {
    const float* x1    = (const float*)d_in[0];
    const float* key_w = (const float*)d_in[1];
    const float* value = (const float*)d_in[2];
    float* out = (float*)d_out;
    (void)in_sizes; (void)n_in; (void)out_size;

    const size_t HALF_BYTES = (size_t)64 * SEQ * HDIM * sizeof(_Float16); // 8 MB
    if (ws_size >= 2 * HALF_BYTES && d_ws != nullptr) {
        _Float16* Kh = (_Float16*)d_ws;
        _Float16* Vt = (_Float16*)((char*)d_ws + HALF_BYTES);
        convert_kv<<<dim3(4096), dim3(256), 0, stream>>>(key_w, value, Kh, Vt);
        attn_fwd_async<<<dim3(64 * 16), dim3(128), 0, stream>>>(x1, Kh, Vt, out);
    } else {
        attn_fwd_f16wmma<<<dim3(64 * 16), dim3(128), 0, stream>>>(x1, key_w, value, out);
    }
}
